// xLSTM_21199958573904
// MI455X (gfx1250) — compile-verified, matrix operand order checked by reading, not run on hardware
//
#include <hip/hip_runtime.h>

// ---------------------------------------------------------------------------
// xLSTM forward for MI455X (gfx1250, wave32, WMMA).
// Dense math via V_WMMA_F32_16X16X32_BF16 (fp32->bf16 tiles, f32 accumulate).
// GEMM: 256x128 block, 64x64 wave tiles, double-buffered LDS (1 barrier/K-step),
// b128 global staging, register prefetch, global_prefetch_b8 two tiles ahead.
// Sequential sLSTM recurrence on VALU (latency bound by definition).
// ---------------------------------------------------------------------------

#define S_TOT   512
#define D_MODEL 1024
#define DI      2048
#define NHEAD   4
#define DHM     512
#define DHS     256
#define FFDIM   1331
#define BATCH   4
#define SMETA   6
#define STOK    506
#define VOCABSZ 8000

typedef unsigned short u16;
typedef unsigned int   u32;
typedef __attribute__((ext_vector_type(16))) __bf16 bf16x16;
typedef __attribute__((ext_vector_type(8)))  float  f32x8;

union Frag { bf16x16 v; u32 u[8]; };

__device__ __forceinline__ u16 f2bf(float f) {
  union { float f; u32 i; } a; a.f = f;
  u32 r = a.i + 0x7fffu + ((a.i >> 16) & 1u);   // round-to-nearest-even
  return (u16)(r >> 16);
}

__device__ __forceinline__ u32 pack2(float a, float b) {
  return (u32)f2bf(a) | ((u32)f2bf(b) << 16);
}

__device__ __forceinline__ f32x8 zero8() {
  f32x8 z = {0.f,0.f,0.f,0.f,0.f,0.f,0.f,0.f};
  return z;
}

__device__ __forceinline__ f32x8 wmma_bf16(bf16x16 a, bf16x16 b, f32x8 c) {
  return __builtin_amdgcn_wmma_f32_16x16x32_bf16(false, a, false, b, (short)0, c,
                                                 false, false);
}

// Load a 16x32 bf16 fragment from an LDS tile laid out [row][k] (k contiguous,
// even stride in u16 units). Same per-lane K mapping serves the A operand
// (row = M) and the B operand (row = N, column-major K runs).
__device__ __forceinline__ bf16x16 load_frag_lds(const u16* tile, int rowbase,
                                                 int stride, int kbase, int lane) {
  Frag f;
  const int r  = rowbase + (lane & 15);
  const int kh = (lane & 16) ? 8 : 0;
  const u16* p = tile + r * stride + kbase + kh;
#pragma unroll
  for (int e = 0; e < 8; ++e) {
    int k = ((e & 4) ? 16 : 0) + ((e & 3) << 1);
    f.u[e] = *(const u32*)(p + k);
  }
  return f.v;
}

// ---------------------------------------------------------------------------
// Generic GEMM: C[M,N] = A[M,K] @ B[K,N] (+bias) with epilogues.
// Block 256x128, K step 32; 8 waves as 4(M) x 2(N), each owns 64x64:
// 16 WMMA per K-step per wave against 8 fragment loads.
// epi: 0 = store (+bias if given), 2 = gelu(acc+bias), 3 = C += acc (+bias).
// slice!=0: row (b,s) remap dropping the SMETA meta rows (vocab head).
// ---------------------------------------------------------------------------
#define BM 256
#define BN 128
#define BK 32
#define LDT 34   // LDS tile stride (u16)

__global__ void __launch_bounds__(256)
gemm_bf16(const float* __restrict__ A, int lda,
          const float* __restrict__ Bm, int ldb,
          float* C, int ldc,
          const float* __restrict__ bias,
          int M, int N, int K, int epi, int slice)
{
  __shared__ u16 As[2][BM * LDT];   // 2 x 17.0 KB
  __shared__ u16 Bs[2][BN * LDT];   // 2 x  8.5 KB
  const int tid  = threadIdx.x;
  const int lane = tid & 31;
  const int wave = tid >> 5;
  const int mw   = wave & 3;    // 4 waves along M (64 rows each)
  const int nw   = wave >> 2;   // 2 waves along N (64 cols each)
  const int m0   = blockIdx.y * BM;
  const int n0   = blockIdx.x * BN;

  const bool avec0 = (m0 + BM <= M) && ((lda & 3) == 0);
  const bool bvec0 = (n0 + BN <= N) && ((ldb & 3) == 0);

  f32x8 acc[4][4];
#pragma unroll
  for (int i = 0; i < 4; ++i)
#pragma unroll
    for (int j = 0; j < 4; ++j) acc[i][j] = zero8();

  float ar[32], br[16];
  const int nk = (K + BK - 1) / BK;

  // ---------------- tile staging helpers (expressed inline) ----------------
  // A regs: vector mapping  idx4=i*256+tid -> row=idx4>>3, k=(idx4&7)*4+c
  //         scalar mapping  idx =i*256+tid -> row=idx>>5,  k=idx&31
  // B regs: vector mapping  idx4=i*256+tid -> kk=idx4>>5,  n=(idx4&31)*4+c
  //         scalar mapping  idx =i*256+tid -> kk=idx>>7,   n=idx&127

#define LOAD_A_TILE(k0, vA)                                                    \
  if (vA) {                                                                    \
    _Pragma("unroll")                                                          \
    for (int i = 0; i < 8; ++i) {                                              \
      int idx4 = i * 256 + tid;                                                \
      int row = idx4 >> 3, k4 = (idx4 & 7) << 2;                               \
      float4 t = *(const float4*)&A[(size_t)(m0 + row) * lda + (k0) + k4];     \
      ar[i*4+0] = t.x; ar[i*4+1] = t.y; ar[i*4+2] = t.z; ar[i*4+3] = t.w;      \
    }                                                                          \
  } else {                                                                     \
    _Pragma("unroll")                                                          \
    for (int i = 0; i < 32; ++i) {                                             \
      int idx = i * 256 + tid;                                                 \
      int gr = m0 + (idx >> 5), gk = (k0) + (idx & 31);                        \
      ar[i] = (gr < M && gk < K) ? A[(size_t)gr * lda + gk] : 0.f;             \
    }                                                                          \
  }

#define LOAD_B_TILE(k0, vB)                                                    \
  if (vB) {                                                                    \
    _Pragma("unroll")                                                          \
    for (int i = 0; i < 4; ++i) {                                              \
      int idx4 = i * 256 + tid;                                                \
      int kk = idx4 >> 5, n4 = (idx4 & 31) << 2;                               \
      float4 t = *(const float4*)&Bm[(size_t)((k0) + kk) * ldb + n0 + n4];     \
      br[i*4+0] = t.x; br[i*4+1] = t.y; br[i*4+2] = t.z; br[i*4+3] = t.w;      \
    }                                                                          \
  } else {                                                                     \
    _Pragma("unroll")                                                          \
    for (int i = 0; i < 16; ++i) {                                             \
      int idx = i * 256 + tid;                                                 \
      int gn = n0 + (idx & 127), gk = (k0) + (idx >> 7);                       \
      br[i] = (gn < N && gk < K) ? Bm[(size_t)gk * ldb + gn] : 0.f;            \
    }                                                                          \
  }

#define STORE_A_TILE(buf, vA)                                                  \
  if (vA) {                                                                    \
    _Pragma("unroll")                                                          \
    for (int i = 0; i < 8; ++i) {                                              \
      int idx4 = i * 256 + tid;                                                \
      int row = idx4 >> 3, k4 = (idx4 & 7) << 2;                               \
      *(u32*)&As[buf][row * LDT + k4]     = pack2(ar[i*4+0], ar[i*4+1]);       \
      *(u32*)&As[buf][row * LDT + k4 + 2] = pack2(ar[i*4+2], ar[i*4+3]);       \
    }                                                                          \
  } else {                                                                     \
    _Pragma("unroll")                                                          \
    for (int i = 0; i < 32; ++i) {                                             \
      int idx = i * 256 + tid;                                                 \
      As[buf][(idx >> 5) * LDT + (idx & 31)] = f2bf(ar[i]);                    \
    }                                                                          \
  }

#define STORE_B_TILE(buf, vB)                                                  \
  if (vB) {                                                                    \
    _Pragma("unroll")                                                          \
    for (int i = 0; i < 4; ++i) {                                              \
      int idx4 = i * 256 + tid;                                                \
      int kk = idx4 >> 5, n4 = (idx4 & 31) << 2;                               \
      Bs[buf][(n4 + 0) * LDT + kk] = f2bf(br[i*4+0]);                          \
      Bs[buf][(n4 + 1) * LDT + kk] = f2bf(br[i*4+1]);                          \
      Bs[buf][(n4 + 2) * LDT + kk] = f2bf(br[i*4+2]);                          \
      Bs[buf][(n4 + 3) * LDT + kk] = f2bf(br[i*4+3]);                          \
    }                                                                          \
  } else {                                                                     \
    _Pragma("unroll")                                                          \
    for (int i = 0; i < 16; ++i) {                                             \
      int idx = i * 256 + tid;                                                 \
      Bs[buf][(idx & 127) * LDT + (idx >> 7)] = f2bf(br[i]);                   \
    }                                                                          \
  }

  // ---- stage tile 0 ----
  {
    const bool vA = avec0 && (BK <= K);
    const bool vB = bvec0 && (BK <= K);
    LOAD_A_TILE(0, vA)
    LOAD_B_TILE(0, vB)
    STORE_A_TILE(0, vA)
    STORE_B_TILE(0, vB)
  }
  __syncthreads();

  for (int kt = 0; kt < nk; ++kt) {
    const int cur = kt & 1;
    const bool more = (kt + 1) < nk;
    bool vA = false, vB = false;

    if (more) {
      const int k0 = (kt + 1) * BK;
      vA = avec0 && (k0 + BK <= K);
      vB = bvec0 && (k0 + BK <= K);
      // warm L2 two tiles ahead (speculative prefetch, dropped if invalid)
      if (kt + 2 < nk) {
        const int k2 = (kt + 2) * BK;
        if (m0 + tid < M && k2 < K)
          __builtin_prefetch(&A[(size_t)(m0 + tid) * lda + k2], 0, 1);
        if (tid < BK && k2 + tid < K)
          __builtin_prefetch(&Bm[(size_t)(k2 + tid) * ldb + n0], 0, 1);
      }
      // prefetch next tile into registers (overlaps with WMMA below)
      LOAD_A_TILE(k0, vA)
      LOAD_B_TILE(k0, vB)
    }

    // ---- compute on LDS[cur]: 16 WMMA per wave ----
    bf16x16 bfr[4];
#pragma unroll
    for (int ni = 0; ni < 4; ++ni)
      bfr[ni] = load_frag_lds(&Bs[cur][0], nw * 64 + ni * 16, LDT, 0, lane);
#pragma unroll
    for (int mi2 = 0; mi2 < 4; ++mi2) {
      bf16x16 af = load_frag_lds(&As[cur][0], mw * 64 + mi2 * 16, LDT, 0, lane);
#pragma unroll
      for (int ni = 0; ni < 4; ++ni)
        acc[mi2][ni] = wmma_bf16(af, bfr[ni], acc[mi2][ni]);
    }

    if (more) {
      const int nxt = cur ^ 1;
      STORE_A_TILE(nxt, vA)
      STORE_B_TILE(nxt, vB)
      __syncthreads();
    }
  }

  // ---- epilogue ----
#pragma unroll
  for (int mi2 = 0; mi2 < 4; ++mi2)
#pragma unroll
    for (int ni = 0; ni < 4; ++ni)
#pragma unroll
      for (int e = 0; e < 8; ++e) {
        int row = m0 + mw * 64 + mi2 * 16 + e + ((lane & 16) ? 8 : 0);
        int col = n0 + nw * 64 + ni * 16 + (lane & 15);
        if (row < M && col < N) {
          float val = acc[mi2][ni][e];
          if (bias) val += bias[col];
          if (epi == 2) {
            float t = val;
            val = 0.5f * t * (1.f + tanhf(0.7978845608f * (t + 0.044715f * t * t * t)));
          }
          if (slice) {
            int b = row >> 9, s = row & 511;
            if (s >= SMETA)
              C[((size_t)(b * STOK + (s - SMETA))) * ldc + col] = val;
          } else {
            size_t ci = (size_t)row * ldc + col;
            if (epi == 3) C[ci] += val; else C[ci] = val;
          }
        }
      }
}

// ---------------------------------------------------------------------------
// Embedding gather
// ---------------------------------------------------------------------------
__global__ void embed_k(const int* __restrict__ tokens, const int* __restrict__ meta,
                        const float* __restrict__ tok_emb, const float* __restrict__ meta_emb,
                        float* __restrict__ x)
{
  const int total = BATCH * S_TOT * D_MODEL;
  for (int idx = blockIdx.x * blockDim.x + threadIdx.x; idx < total;
       idx += gridDim.x * blockDim.x) {
    int d = idx & 1023, bs = idx >> 10;
    int s = bs & 511, b = bs >> 9;
    if (s < SMETA) x[idx] = meta_emb[(size_t)meta[b * SMETA + s] * D_MODEL + d];
    else           x[idx] = tok_emb[(size_t)tokens[b * STOK + (s - SMETA)] * D_MODEL + d];
  }
}

// ---------------------------------------------------------------------------
// LayerNorm (biased variance, eps 1e-5), one row per block
// ---------------------------------------------------------------------------
__global__ void __launch_bounds__(256)
layernorm_k(const float* __restrict__ in, const float* __restrict__ w,
            float* __restrict__ out, int d)
{
  __shared__ float red[256];
  __shared__ float stats[2];
  const int row = blockIdx.x, tid = threadIdx.x;
  const float* xr = in + (size_t)row * d;
  float s = 0.f;
  for (int i = tid; i < d; i += 256) s += xr[i];
  red[tid] = s; __syncthreads();
  for (int o = 128; o > 0; o >>= 1) { if (tid < o) red[tid] += red[tid + o]; __syncthreads(); }
  if (tid == 0) stats[0] = red[0] / d;
  __syncthreads();
  float mean = stats[0], s2 = 0.f;
  for (int i = tid; i < d; i += 256) { float t = xr[i] - mean; s2 += t * t; }
  red[tid] = s2; __syncthreads();
  for (int o = 128; o > 0; o >>= 1) { if (tid < o) red[tid] += red[tid + o]; __syncthreads(); }
  if (tid == 0) stats[1] = red[0] / d;
  __syncthreads();
  float rstd = rsqrtf(stats[1] + 1e-5f);
  for (int i = tid; i < d; i += 256)
    out[(size_t)row * d + i] = (xr[i] - mean) * rstd * w[i];
}

// ---------------------------------------------------------------------------
// Causal depthwise conv (K=4) + SiLU; in_stride handles the xi view of `up`
// ---------------------------------------------------------------------------
__global__ void conv_silu_k(const float* __restrict__ in, int in_stride,
                            const float* __restrict__ w,
                            float* __restrict__ out, int C)
{
  const int total = BATCH * S_TOT * C;
  for (int idx = blockIdx.x * blockDim.x + threadIdx.x; idx < total;
       idx += gridDim.x * blockDim.x) {
    int c = idx % C, bs = idx / C;
    int s = bs % S_TOT, b = bs / S_TOT;
    float a = 0.f;
#pragma unroll
    for (int j = 0; j < 4; ++j) {
      int sp = s + j - 3;
      if (sp >= 0) a += in[((size_t)(b * S_TOT + sp)) * in_stride + c] * w[c * 4 + j];
    }
    out[idx] = a / (1.f + __expf(-a));   // silu
  }
}

// ---------------------------------------------------------------------------
// Block-diagonal (512 blocks of 4x4) projections for q (ca), k (ca), v (xi)
// ---------------------------------------------------------------------------
__global__ void blocklin_k(const float* __restrict__ ca, const float* __restrict__ up,
                           const float* __restrict__ W,
                           float* __restrict__ q, float* __restrict__ k2,
                           float* __restrict__ v2)
{
  const int total = 3 * BATCH * S_TOT * 512;
  for (int idx = blockIdx.x * blockDim.x + threadIdx.x; idx < total;
       idx += gridDim.x * blockDim.x) {
    int n = idx % 512, rem = idx / 512;
    int row = rem % (BATCH * S_TOT), t = rem / (BATCH * S_TOT);
    const float* inr = (t == 2) ? (up + (size_t)row * 4096) : (ca + (size_t)row * DI);
    float x0 = inr[n * 4 + 0], x1 = inr[n * 4 + 1], x2 = inr[n * 4 + 2], x3 = inr[n * 4 + 3];
    const float* Wt = W + ((size_t)t * 512 + n) * 16;
    float* dst = (t == 0) ? q : (t == 1) ? k2 : v2;
#pragma unroll
    for (int jj = 0; jj < 4; ++jj)
      dst[(size_t)row * DI + n * 4 + jj] =
          x0 * Wt[jj] + x1 * Wt[4 + jj] + x2 * Wt[8 + jj] + x3 * Wt[12 + jj];
  }
}

// ---------------------------------------------------------------------------
// g = [q,k,v] @ Wif + bif  (N=8, one row per block, tree reduction)
// ---------------------------------------------------------------------------
__global__ void __launch_bounds__(256)
wif_k(const float* __restrict__ q, const float* __restrict__ k2,
      const float* __restrict__ v2, const float* __restrict__ Wif,
      const float* __restrict__ bif, float* __restrict__ g)
{
  __shared__ float red[8][256];
  const int row = blockIdx.x, tid = threadIdx.x;
  float acc[8] = {0.f,0.f,0.f,0.f,0.f,0.f,0.f,0.f};
  for (int i = tid; i < DI; i += 256) {
    float qv = q[(size_t)row * DI + i];
    float kv = k2[(size_t)row * DI + i];
    float vv = v2[(size_t)row * DI + i];
#pragma unroll
    for (int c = 0; c < 8; ++c)
      acc[c] += qv * Wif[i * 8 + c] + kv * Wif[(DI + i) * 8 + c] +
                vv * Wif[(2 * DI + i) * 8 + c];
  }
#pragma unroll
  for (int c = 0; c < 8; ++c) red[c][tid] = acc[c];
  __syncthreads();
  for (int o = 128; o > 0; o >>= 1) {
    if (tid < o) {
#pragma unroll
      for (int c = 0; c < 8; ++c) red[c][tid] += red[c][tid + o];
    }
    __syncthreads();
  }
  if (tid < 8) g[row * 8 + tid] = red[tid][0] + bif[tid];
}

// ---------------------------------------------------------------------------
// Per-(b,h) gate prep: cf = cumsum(logsigmoid(f)); a[t] = i[t]-cf[t];
// pm[s] = prefixmax(a); m[s] = cf[s]+pm[s].  (Exact row max, no online pass.)
// ---------------------------------------------------------------------------
__global__ void gate_prep_k(const float* __restrict__ g, float* __restrict__ abuf,
                            float* __restrict__ pmbuf, float* __restrict__ mbuf)
{
  const int hh = blockIdx.x, bb = blockIdx.y;
  if (threadIdx.x != 0) return;
  float cf = 0.f, pm = -1e30f;
  const int base = (bb * NHEAD + hh) * S_TOT;
  for (int s = 0; s < S_TOT; ++s) {
    int gi = (bb * S_TOT + s) * 8;
    float iv = g[gi + hh];
    float fv = g[gi + 4 + hh];
    float lf = fminf(fv, 0.f) - log1pf(__expf(-fabsf(fv)));
    cf += lf;
    float av = iv - cf;
    pm = fmaxf(pm, av);
    abuf[base + s] = av;
    pmbuf[base + s] = pm;
    mbuf[base + s] = cf + pm;
  }
}

// ---------------------------------------------------------------------------
// mLSTM cell, flash-style. Block = 16 query rows of one (b,h).
// Scores: split-K over 8 waves (WMMA) + LDS reduce; decay/mask in f32;
// out accumulation: each wave owns 64 of the 512 head channels (4 WMMA tiles).
// ---------------------------------------------------------------------------
#define QS_STR 520
#define VT_STR 40
#define ATTN_SMEM_BYTES ((16*QS_STR + 32*QS_STR + 512*VT_STR + 16*VT_STR) * 2 + \
                         (8*512 + 512 + 48) * 4)

__global__ void __launch_bounds__(256)
mlstm_attn(const float* __restrict__ q, const float* __restrict__ k,
           const float* __restrict__ v, const float* __restrict__ abuf,
           const float* __restrict__ pmbuf, const float* __restrict__ mbuf,
           float* __restrict__ hout)
{
  extern __shared__ char smem_raw[];
  u16* Qs = (u16*)smem_raw;            // 16  x 520 (q tile, bf16)
  u16* Ks = Qs + 16 * QS_STR;          // 32  x 520 (keys, K=d)
  u16* Vt = Ks + 32 * QS_STR;          // 512 x 40  (v transposed, K=t)
  u16* Pb = Vt + 512 * VT_STR;         // 16  x 40  (P tile, bf16)
  float* Pp   = (float*)(Pb + 16 * VT_STR);  // 8 x 16 x 32 partial scores
  float* Pval = Pp + 8 * 512;                // 512 weighted scores
  float* rs   = Pval + 512;                  // 16 row sums / norms
  float* pmr  = rs + 16;
  float* mr   = pmr + 16;

  const int tid = threadIdx.x, lane = tid & 31, wv = tid >> 5;
  const int itile = blockIdx.x, hh = blockIdx.y, bb = blockIdx.z;
  const int s0 = itile * 16;
  const size_t qk_base = (size_t)bb * S_TOT * DI + (size_t)hh * DHM;
  const int am_base = (bb * NHEAD + hh) * S_TOT;
  const float inv_sqrt = 0.044194173824159216f;  // 1/sqrt(512)

  // stage Q tile with b128 loads (always 16B aligned: DI stride, d4 multiple of 4)
  for (int i = tid; i < 16 * 128; i += 256) {
    int r = i >> 7, d4 = (i & 127) << 2;
    float4 t = *(const float4*)&q[qk_base + (size_t)(s0 + r) * DI + d4];
    *(u32*)&Qs[r * QS_STR + d4]     = pack2(t.x, t.y);
    *(u32*)&Qs[r * QS_STR + d4 + 2] = pack2(t.z, t.w);
  }
  if (tid < 16) {
    rs[tid]  = 0.f;
    pmr[tid] = pmbuf[am_base + s0 + tid];
    mr[tid]  = mbuf[am_base + s0 + tid];
  }

  f32x8 oacc[4];
#pragma unroll
  for (int ni = 0; ni < 4; ++ni) oacc[ni] = zero8();

  const int jmax = (s0 + 15) >> 5;
  for (int j = 0; j <= jmax; ++j) {
    const int t0 = j << 5;
    __syncthreads();
    for (int i = tid; i < 32 * 128; i += 256) {
      int t = i >> 7, d4 = (i & 127) << 2;
      size_t gi = qk_base + (size_t)(t0 + t) * DI + d4;
      float4 kv = *(const float4*)&k[gi];
      float4 vv = *(const float4*)&v[gi];
      *(u32*)&Ks[t * QS_STR + d4]     = pack2(kv.x, kv.y);
      *(u32*)&Ks[t * QS_STR + d4 + 2] = pack2(kv.z, kv.w);
      Vt[(d4 + 0) * VT_STR + t] = f2bf(vv.x);
      Vt[(d4 + 1) * VT_STR + t] = f2bf(vv.y);
      Vt[(d4 + 2) * VT_STR + t] = f2bf(vv.z);
      Vt[(d4 + 3) * VT_STR + t] = f2bf(vv.w);
    }
    __syncthreads();

    // raw scores: wave wv handles feature chunk [wv*64, wv*64+64)
    f32x8 sc0 = zero8(), sc1 = zero8();
#pragma unroll
    for (int kb = 0; kb < 2; ++kb) {
      int kbase = wv * 64 + kb * 32;
      bf16x16 aq = load_frag_lds(Qs, 0,  QS_STR, kbase, lane);
      bf16x16 b0 = load_frag_lds(Ks, 0,  QS_STR, kbase, lane);
      bf16x16 b1 = load_frag_lds(Ks, 16, QS_STR, kbase, lane);
      sc0 = wmma_bf16(aq, b0, sc0);
      sc1 = wmma_bf16(aq, b1, sc1);
    }
#pragma unroll
    for (int e = 0; e < 8; ++e) {
      int r = e + ((lane & 16) ? 8 : 0);
      Pp[wv * 512 + r * 32 + (lane & 15)]      = sc0[e];
      Pp[wv * 512 + r * 32 + 16 + (lane & 15)] = sc1[e];
    }
    __syncthreads();

    // reduce partials, apply decay weights + causal mask
    for (int e2 = tid; e2 < 512; e2 += 256) {
      int r = e2 >> 5, c = e2 & 31;
      float sum = 0.f;
#pragma unroll
      for (int w2 = 0; w2 < 8; ++w2) sum += Pp[w2 * 512 + e2];
      int sIdx = s0 + r, tIdx = t0 + c;
      float val = 0.f;
      if (tIdx <= sIdx)
        val = sum * inv_sqrt * __expf(abuf[am_base + tIdx] - pmr[r]);
      Pval[e2] = val;
      Pb[r * VT_STR + c] = f2bf(val);
    }
    __syncthreads();
    if (tid < 16) {
      float a2 = rs[tid];
#pragma unroll
      for (int c = 0; c < 32; ++c) a2 += Pval[tid * 32 + c];
      rs[tid] = a2;
    }
    // out += P(16x32) @ V(32x64-chunk)
    bf16x16 pa = load_frag_lds(Pb, 0, VT_STR, 0, lane);
#pragma unroll
    for (int ni = 0; ni < 4; ++ni) {
      bf16x16 bv = load_frag_lds(Vt, wv * 64 + ni * 16, VT_STR, 0, lane);
      oacc[ni] = wmma_bf16(pa, bv, oacc[ni]);
    }
  }
  __syncthreads();
  if (tid < 16) rs[tid] = fmaxf(fabsf(rs[tid]), __expf(-mr[tid]));
  __syncthreads();
#pragma unroll
  for (int ni = 0; ni < 4; ++ni)
#pragma unroll
    for (int e = 0; e < 8; ++e) {
      int r = e + ((lane & 16) ? 8 : 0);
      int col = hh * DHM + wv * 64 + ni * 16 + (lane & 15);
      hout[((size_t)(bb * S_TOT + s0 + r)) * DI + col] = oacc[ni][e] / rs[r];
    }
}

// ---------------------------------------------------------------------------
// mLSTM epilogue: y = GN(h + skip*ca, NH groups of 512) * gnw * silu(z)
// (in-place safe; z lives in the upper half of `up`)
// ---------------------------------------------------------------------------
__global__ void __launch_bounds__(256)
gn_silu_k(const float* h, const float* __restrict__ ca,
          const float* __restrict__ up, const float* __restrict__ skip,
          const float* __restrict__ gnw, float* out)
{
  __shared__ float buf[DI];
  __shared__ float red[256];
  __shared__ float stats[2];
  const int row = blockIdx.x, tid = threadIdx.x;
  for (int e = tid; e < DI; e += 256)
    buf[e] = h[(size_t)row * DI + e] + skip[e] * ca[(size_t)row * DI + e];
  __syncthreads();
  for (int g = 0; g < 4; ++g) {
    float s = 0.f;
    for (int e = g * 512 + tid; e < g * 512 + 512; e += 256) s += buf[e];
    red[tid] = s; __syncthreads();
    for (int o = 128; o > 0; o >>= 1) { if (tid < o) red[tid] += red[tid + o]; __syncthreads(); }
    if (tid == 0) stats[0] = red[0] * (1.f / 512.f);
    __syncthreads();
    float mean = stats[0];
    s = 0.f;
    for (int e = g * 512 + tid; e < g * 512 + 512; e += 256) {
      float t = buf[e] - mean; s += t * t;
    }
    red[tid] = s; __syncthreads();
    for (int o = 128; o > 0; o >>= 1) { if (tid < o) red[tid] += red[tid + o]; __syncthreads(); }
    if (tid == 0) stats[1] = red[0] * (1.f / 512.f);
    __syncthreads();
    float rstd = rsqrtf(stats[1] + 1e-5f);
    for (int e = g * 512 + tid; e < g * 512 + 512; e += 256) {
      float zn = up[(size_t)row * 4096 + DI + e];
      out[(size_t)row * DI + e] =
          (buf[e] - mean) * rstd * gnw[e] * (zn / (1.f + __expf(-zn)));
    }
    __syncthreads();
  }
}

// ---------------------------------------------------------------------------
// sLSTM epilogue: x += GN(h, NH groups of 256) * gnw
// ---------------------------------------------------------------------------
__global__ void __launch_bounds__(256)
gn_resid_k(const float* __restrict__ h, const float* __restrict__ gnw, float* x)
{
  __shared__ float red[256];
  __shared__ float stats[2];
  const int row = blockIdx.x, tid = threadIdx.x;
  for (int g = 0; g < 4; ++g) {
    int e = g * 256 + tid;
    float v = h[(size_t)row * D_MODEL + e];
    red[tid] = v; __syncthreads();
    for (int o = 128; o > 0; o >>= 1) { if (tid < o) red[tid] += red[tid + o]; __syncthreads(); }
    if (tid == 0) stats[0] = red[0] * (1.f / 256.f);
    __syncthreads();
    float mean = stats[0];
    float t2 = v - mean;
    red[tid] = t2 * t2; __syncthreads();
    for (int o = 128; o > 0; o >>= 1) { if (tid < o) red[tid] += red[tid + o]; __syncthreads(); }
    if (tid == 0) stats[1] = red[0] * (1.f / 256.f);
    __syncthreads();
    float rstd = rsqrtf(stats[1] + 1e-5f);
    x[(size_t)row * D_MODEL + e] += (v - mean) * rstd * gnw[e];
    __syncthreads();
  }
}

// ---------------------------------------------------------------------------
// sLSTM recurrence. One WG per head; state (c,n,m) in registers per thread
// (thread = output channel e), h broadcast via LDS; 512 sequential steps.
// ---------------------------------------------------------------------------
__global__ void __launch_bounds__(256)
slstm_scan_k(const float* __restrict__ pre, const float* __restrict__ R,
             float* __restrict__ hs)
{
  const int n = blockIdx.x, e = threadIdx.x;
  __shared__ float hsh[4][DHS];
  float c[4]  = {0.f,0.f,0.f,0.f};
  float nn[4] = {0.f,0.f,0.f,0.f};
  float m[4]  = {0.f,0.f,0.f,0.f};
#pragma unroll
  for (int b = 0; b < 4; ++b) hsh[b][e] = 0.f;
  __syncthreads();
  const size_t gstride = (size_t)BATCH * S_TOT * D_MODEL;
  for (int s = 0; s < S_TOT; ++s) {
    float r[4][4];
#pragma unroll
    for (int g = 0; g < 4; ++g)
#pragma unroll
      for (int b = 0; b < 4; ++b) r[g][b] = 0.f;
    for (int d = 0; d < DHS; ++d) {
      float h0 = hsh[0][d], h1 = hsh[1][d], h2 = hsh[2][d], h3 = hsh[3][d];
#pragma unroll
      for (int g = 0; g < 4; ++g) {
        float rv = R[(((size_t)g * 4 + n) * DHS + d) * DHS + e];
        r[g][0] += rv * h0; r[g][1] += rv * h1;
        r[g][2] += rv * h2; r[g][3] += rv * h3;
      }
    }
    __syncthreads();
#pragma unroll
    for (int b = 0; b < 4; ++b) {
      size_t pb = ((size_t)b * S_TOT + s) * D_MODEL + n * DHS + e;
      float pz = pre[0 * gstride + pb] + r[0][b];
      float pi = pre[1 * gstride + pb] + r[1][b];
      float pf = pre[2 * gstride + pb] + r[2][b];
      float po = pre[3 * gstride + pb] + r[3][b];
      float zt = tanhf(pz);
      float ot = 1.f / (1.f + __expf(-po));
      float lf = fminf(pf, 0.f) - log1pf(__expf(-fabsf(pf)));
      float mn = fmaxf(lf + m[b], pi);
      float i_ = __expf(pi - mn);
      float f_ = __expf(lf + m[b] - mn);
      c[b]  = f_ * c[b] + i_ * zt;
      nn[b] = f_ * nn[b] + i_;
      float hv = ot * (c[b] / nn[b]);
      m[b] = mn;
      hs[pb] = hv;
      hsh[b][e] = hv;
    }
    __syncthreads();
  }
}

// ---------------------------------------------------------------------------
// Host orchestration
// ---------------------------------------------------------------------------
extern "C" void kernel_launch(void* const* d_in, const int* in_sizes, int n_in,
                              void* d_out, int out_size, void* d_ws, size_t ws_size,
                              hipStream_t stream)
{
  const int*   tokens   = (const int*)d_in[0];
  const int*   meta     = (const int*)d_in[1];
  const float* tok_emb  = (const float*)d_in[2];
  const float* meta_emb = (const float*)d_in[3];
  const float* m_ln     = (const float*)d_in[4];
  const float* m_Wup    = (const float*)d_in[5];
  const float* m_conv   = (const float*)d_in[6];
  const float* m_Wqkv   = (const float*)d_in[7];
  const float* m_Wif    = (const float*)d_in[8];
  const float* m_bif    = (const float*)d_in[9];
  const float* m_skip   = (const float*)d_in[10];
  const float* m_gn     = (const float*)d_in[11];
  const float* m_Wdown  = (const float*)d_in[12];
  const float* s_ln     = (const float*)d_in[13];
  const float* s_conv   = (const float*)d_in[14];
  const float* s_W      = (const float*)d_in[15];
  const float* s_R      = (const float*)d_in[16];
  const float* s_b      = (const float*)d_in[17];
  const float* s_gn     = (const float*)d_in[18];
  const float* s_ln2    = (const float*)d_in[19];
  const float* s_W1     = (const float*)d_in[20];
  const float* s_b1     = (const float*)d_in[21];
  const float* s_W2     = (const float*)d_in[22];
  const float* s_b2     = (const float*)d_in[23];
  const float* final_ln = (const float*)d_in[24];
  const float* out_W    = (const float*)d_in[25];
  const float* out_b    = (const float*)d_in[26];
  float* out = (float*)d_out;
  (void)in_sizes; (void)n_in; (void)out_size; (void)ws_size;

  const size_t ROWS = (size_t)BATCH * S_TOT;  // 2048
  float* ws  = (float*)d_ws;
  float* x   = ws;  ws += ROWS * D_MODEL;
  float* xn  = ws;  ws += ROWS * D_MODEL;
  float* up  = ws;  ws += ROWS * 4096;      // mLSTM up-proj / sLSTM pre-gates
  float* ca  = ws;  ws += ROWS * DI;
  float* qb  = ws;  ws += ROWS * DI;        // also sLSTM hidden states
  float* kb  = ws;  ws += ROWS * DI;        // also sLSTM FFN hidden
  float* vb  = ws;  ws += ROWS * DI;
  float* hb  = ws;  ws += ROWS * DI;
  float* gb  = ws;  ws += ROWS * 8;
  float* ab  = ws;  ws += BATCH * NHEAD * S_TOT;
  float* pmb = ws;  ws += BATCH * NHEAD * S_TOT;
  float* mb  = ws;  ws += BATCH * NHEAD * S_TOT;

  embed_k<<<8192, 256, 0, stream>>>(tokens, meta, tok_emb, meta_emb, x);

  int mi = 0, si = 0;
  for (int blk = 0; blk < 11; ++blk) {
    const bool is_s = (blk == 1 || blk == 4 || blk == 7 || blk == 10);
    if (!is_s) {
      layernorm_k<<<2048, 256, 0, stream>>>(x, m_ln + (size_t)mi * D_MODEL, xn, D_MODEL);
      gemm_bf16<<<dim3(32, 8), 256, 0, stream>>>(
          xn, D_MODEL, m_Wup + (size_t)mi * D_MODEL * 4096, 4096,
          up, 4096, nullptr, 2048, 4096, 1024, 0, 0);
      conv_silu_k<<<(int)((ROWS * DI + 255) / 256), 256, 0, stream>>>(
          up, 4096, m_conv + (size_t)mi * DI * 4, ca, DI);
      blocklin_k<<<12288, 256, 0, stream>>>(ca, up, m_Wqkv + (size_t)mi * 3 * 512 * 16,
                                            qb, kb, vb);
      wif_k<<<2048, 256, 0, stream>>>(qb, kb, vb, m_Wif + (size_t)mi * 6144 * 8,
                                      m_bif + mi * 8, gb);
      gate_prep_k<<<dim3(NHEAD, BATCH), 32, 0, stream>>>(gb, ab, pmb, mb);
      mlstm_attn<<<dim3(32, NHEAD, BATCH), 256, ATTN_SMEM_BYTES, stream>>>(
          qb, kb, vb, ab, pmb, mb, hb);
      gn_silu_k<<<2048, 256, 0, stream>>>(hb, ca, up, m_skip + (size_t)mi * DI,
                                          m_gn + (size_t)mi * DI, hb);
      gemm_bf16<<<dim3(8, 8), 256, 0, stream>>>(
          hb, DI, m_Wdown + (size_t)mi * DI * D_MODEL, D_MODEL,
          x, D_MODEL, nullptr, 2048, 1024, 2048, 3, 0);
      ++mi;
    } else {
      layernorm_k<<<2048, 256, 0, stream>>>(x, s_ln + (size_t)si * D_MODEL, xn, D_MODEL);
      conv_silu_k<<<(int)((ROWS * D_MODEL + 255) / 256), 256, 0, stream>>>(
          xn, D_MODEL, s_conv + (size_t)si * D_MODEL * 4, ca, D_MODEL);
      for (int g = 0; g < 4; ++g) {
        const float* Ain = (g == 1 || g == 2) ? ca : xn;   // i,f use conv path
        for (int nh = 0; nh < 4; ++nh) {
          gemm_bf16<<<dim3(2, 8), 256, 0, stream>>>(
              Ain + nh * DHS, D_MODEL,
              s_W + ((size_t)(si * 4 + g) * 4 + nh) * DHS * DHS, DHS,
              up + (size_t)g * ROWS * D_MODEL + nh * DHS, D_MODEL,
              s_b + (size_t)si * 4 * D_MODEL + g * D_MODEL + nh * DHS,
              2048, DHS, DHS, 0, 0);
        }
      }
      slstm_scan_k<<<NHEAD, 256, 0, stream>>>(up, s_R + (size_t)si * 4 * NHEAD * DHS * DHS, qb);
      gn_resid_k<<<2048, 256, 0, stream>>>(qb, s_gn + (size_t)si * D_MODEL, x);
      layernorm_k<<<2048, 256, 0, stream>>>(x, s_ln2 + (size_t)si * D_MODEL, xn, D_MODEL);
      gemm_bf16<<<dim3(11, 8), 256, 0, stream>>>(
          xn, D_MODEL, s_W1 + (size_t)si * D_MODEL * FFDIM, FFDIM,
          kb, FFDIM, s_b1 + (size_t)si * FFDIM, 2048, FFDIM, 1024, 2, 0);
      gemm_bf16<<<dim3(8, 8), 256, 0, stream>>>(
          kb, FFDIM, s_W2 + (size_t)si * FFDIM * D_MODEL, D_MODEL,
          x, D_MODEL, s_b2 + (size_t)si * D_MODEL, 2048, 1024, FFDIM, 3, 0);
      ++si;
    }
  }
  layernorm_k<<<2048, 256, 0, stream>>>(x, final_ln, xn, D_MODEL);
  gemm_bf16<<<dim3(63, 8), 256, 0, stream>>>(
      xn, D_MODEL, out_W, VOCABSZ, out, VOCABSZ, out_b,
      2048, VOCABSZ, 1024, 0, 1);
}
